// Attention_57913339020018
// MI455X (gfx1250) — compile-verified, matrix operand order checked by reading, not run on hardware
//
#include <hip/hip_runtime.h>
#include <hip/hip_bf16.h>
#include <math.h>

// Problem constants (reference: B=16, S=2048, H=1024)
#define Hd 1024
#define Sd 2048
#define Bd 16
#define TMR 64                // s-rows per block (four 16-row WMMA M-subtiles)

typedef __attribute__((ext_vector_type(16))) __bf16   v16bf;
typedef __attribute__((ext_vector_type(8)))  float    v8f;
typedef __attribute__((ext_vector_type(8)))  unsigned v8u;

__device__ __forceinline__ unsigned pack2_bf16_rne(float a, float b) {
  unsigned ua = __builtin_bit_cast(unsigned, a);
  unsigned ub = __builtin_bit_cast(unsigned, b);
  ua += 0x7FFFu + ((ua >> 16) & 1u);   // RNE round to bf16
  ub += 0x7FFFu + ((ub >> 16) & 1u);
  return (ua >> 16) | (ub & 0xFFFF0000u);
}

__device__ __forceinline__ float fast_tanh(float x) {
#if __has_builtin(__builtin_amdgcn_tanhf)
  return __builtin_amdgcn_tanhf(x);     // native v_tanh_f32 on gfx1250
#else
  return tanhf(x);
#endif
}

// Kernel 0: pre-pack W2 = W_attn[H:] into bf16 with a B-fragment-native tiling:
//   Wq[(kp>>2)*H + n] is a uint4 holding K-pairs kp..kp+3 of column n.
__global__ __launch_bounds__(256) void packw_kernel(
    const float* __restrict__ W, unsigned* __restrict__ Wpk) {
  int idx = blockIdx.x * 256 + threadIdx.x;       // 512*1024 output dwords
  int kpo = idx & 3;                              // K-pair within group of 4
  int n   = (idx >> 2) & 1023;                    // column
  int kpg = idx >> 12;                            // K-pair group (0..127)
  int k   = ((kpg << 2) | kpo) * 2;
  const float* W2 = W + (size_t)Hd * Hd;
  float w0 = W2[(size_t)k * Hd + n];
  float w1 = W2[(size_t)(k + 1) * Hd + n];
  Wpk[idx] = pack2_bf16_rne(w0, w1);
}

// Kernel 1: hidb[b][k] = hidden[b] . W_attn[:H][:,k] + b_attn[k]   (fp32)
__global__ __launch_bounds__(256) void hid_kernel(
    const float* __restrict__ hidden, const float* __restrict__ W,
    const float* __restrict__ bias, float* __restrict__ hidb) {
  int idx = blockIdx.x * 256 + threadIdx.x;       // 16*1024 outputs
  int bb = idx >> 10, kk = idx & 1023;
  const float* hrow = hidden + (size_t)bb * Hd;
  float s = bias[kk];
  for (int j = 0; j < Hd; ++j)
    s = fmaf(hrow[j], W[(size_t)j * Hd + kk], s);
  hidb[idx] = s;
}

// Kernel 2: fused  logits[b][s] = sum_n tanh( (enc @ W2)[s][n] + hidb[b][n] ) * v[n]
// One block = 64 s-rows of one batch (four M-subtiles), 128KB A tile in LDS
// (CDNA5 WGP allows up to 320KB per workgroup). 8 waves; each wave covers 8
// n-tiles in 2 groups of 4, holding C[4 n-tiles][4 subtiles]; every B fragment
// is amortized over 4 WMMAs -> W L2 traffic halves vs the 32-row tile.
__global__ __launch_bounds__(256) void attn_energy_kernel(
    const float* __restrict__ enc, const unsigned* __restrict__ Wpk,
    const float* __restrict__ hidb, const float* __restrict__ vvec,
    float* __restrict__ logits) {
  __shared__ unsigned ldsA[512 * TMR];   // 128KB (reused as f32 scratch later)

  const int tid  = threadIdx.x;
  const int wave = tid >> 5;
  const int lane = tid & 31;
  const int b    = blockIdx.x >> 5;            // Sd/TMR = 32 tiles per batch
  const int s0   = (blockIdx.x & 31) * TMR;

  // Stage A tile: 64 rows x 1024 K, fp32 -> packed bf16 pairs, XOR-swizzled:
  //   dword idx = row*512 + (kp ^ ((row&15)<<2))
  const float4* encv = (const float4*)(enc + ((size_t)b * Sd + s0) * Hd);
  for (int i = 0; i < 64; ++i) {
    int c   = tid + (i << 8);                  // float4 chunk id, 0..16383
    int row = c >> 8;                          // 256 float4 per row
    int kq  = c & 255;
    float4 a = encv[(size_t)row * 256 + kq];
    int kp0 = 2 * kq;
    int s   = (row & 15) << 2;
    uint2 pr;
    pr.x = pack2_bf16_rne(a.x, a.y);
    pr.y = pack2_bf16_rne(a.z, a.w);
    *(uint2*)&ldsA[row * 512 + (kp0 ^ s)] = pr;
  }
  __syncthreads();

  // WMMA fragment addressing (ISA 7.12.2, wave32):
  //  A (16x32 bf16): lane&15 = M row; lane>=16 -> K-pair offset 4;
  //                  VGPR j holds K-pair kp = 16*ks + (j&4)*2 + 4*hi + (j&3).
  //  B (32x16 bf16): lane&15 = N col; same K-pair pattern.
  //  C (16x16 f32):  VGPR r -> row r + 8*(lane>>4), col = lane&15.
  const int rowa = lane & 15;                  // A row == B column within tile
  const int hi   = lane >> 4;
  const int swz  = rowa << 2;                  // XOR swizzle for this lane's row
  const uint4* Wq = (const uint4*)Wpk;         // [kpg*Hd + n]

  float acc[4][8];
#pragma unroll
  for (int sub = 0; sub < 4; ++sub)
#pragma unroll
    for (int r = 0; r < 8; ++r) acc[sub][r] = 0.f;

  for (int g = 0; g < 2; ++g) {                // 2 groups of 4 n-tiles per wave
    const int nbase = wave * 128 + g * 64 + rowa;
    v8f C[4][4];                               // [n-tile][M-subtile]
#pragma unroll
    for (int t = 0; t < 4; ++t)
#pragma unroll
      for (int sub = 0; sub < 4; ++sub) C[t][sub] = (v8f){};

    // B pointer for ks=0: kpg = (ks<<2)+hi -> advance by 4*Hd uint4 per ks.
    const uint4* pB = Wq + (size_t)hi * Hd + nbase;

    for (int ks = 0; ks < 32; ++ks) {
      const int base0 = (ks << 4) + (hi << 2);        // K-pair base, jh=0
      const int col0  = base0 ^ swz;
      const int col1  = (base0 + 8) ^ swz;
      // A fragments once per ks: 2x ds_load_b128 per M-subtile (4 subtiles).
      v16bf Af[4];
#pragma unroll
      for (int sub = 0; sub < 4; ++sub) {
        const int rbase = (sub * 16 + rowa) * 512;
        uint4 alo = *(const uint4*)&ldsA[rbase + col0];
        uint4 ahi = *(const uint4*)&ldsA[rbase + col1];
        v8u au = {alo.x, alo.y, alo.z, alo.w, ahi.x, ahi.y, ahi.z, ahi.w};
        Af[sub] = __builtin_bit_cast(v16bf, au);
      }

#pragma unroll
      for (int t = 0; t < 4; ++t) {                   // 4 n-tiles share A-frags
        uint4 blo = pB[t * 16];                       // +256B immediate offsets
        uint4 bhi = pB[2 * Hd + t * 16];
        v8u bu = {blo.x, blo.y, blo.z, blo.w, bhi.x, bhi.y, bhi.z, bhi.w};
        v16bf Bm = __builtin_bit_cast(v16bf, bu);
#pragma unroll
        for (int sub = 0; sub < 4; ++sub)             // B amortized over 4 subs
          C[t][sub] = __builtin_amdgcn_wmma_f32_16x16x32_bf16(
                          false, Af[sub], false, Bm, (short)0, C[t][sub],
                          false, false);
      }
      pB += 4 * Hd;                                   // next ks K-pair group
    }

    // Fused epilogue: tanh(C + hid_part + bias) * v[n], reduce over n per row.
#pragma unroll
    for (int t = 0; t < 4; ++t) {
      const int n = nbase + t * 16;
      const float hb = hidb[b * Hd + n];
      const float vn = vvec[n];
#pragma unroll
      for (int sub = 0; sub < 4; ++sub)
#pragma unroll
        for (int r = 0; r < 8; ++r)
          acc[sub][r] += fast_tanh(C[t][sub][r] + hb) * vn;  // row = sub*16 + r + 8*hi
    }
  }

  // Deterministic cross-lane/cross-wave reduction: reuse A-tile LDS (A is dead).
  __syncthreads();
  float* scratch = (float*)ldsA;                      // 256 threads x 32 floats
#pragma unroll
  for (int sub = 0; sub < 4; ++sub)
#pragma unroll
    for (int r = 0; r < 8; ++r)
      scratch[tid * 32 + sub * 8 + r] = acc[sub][r];
  __syncthreads();
  if (tid < TMR) {
    const int sub = tid >> 4, h = (tid >> 3) & 1, r = tid & 7;
    float s = 0.f;
    for (int w = 0; w < 8; ++w)
      for (int l = 0; l < 16; ++l)
        s += scratch[(w * 32 + h * 16 + l) * 32 + sub * 8 + r];
    logits[(size_t)b * Sd + s0 + tid] = s;
  }
}

// Kernel 3: softmax over S per batch row.
__global__ __launch_bounds__(256) void softmax_kernel(
    const float* __restrict__ logits, float* __restrict__ out) {
  __shared__ float red[256];
  const int b = blockIdx.x, tid = threadIdx.x;
  const float* row = logits + (size_t)b * Sd;

  float m = -INFINITY;
  for (int i = tid; i < Sd; i += 256) m = fmaxf(m, row[i]);
  red[tid] = m; __syncthreads();
  for (int s = 128; s > 0; s >>= 1) {
    if (tid < s) red[tid] = fmaxf(red[tid], red[tid + s]);
    __syncthreads();
  }
  m = red[0];
  __syncthreads();

  float sl = 0.f;
  for (int i = tid; i < Sd; i += 256) {
    float e = __expf(row[i] - m);
    out[(size_t)b * Sd + i] = e;
    sl += e;
  }
  red[tid] = sl; __syncthreads();
  for (int s = 128; s > 0; s >>= 1) {
    if (tid < s) red[tid] += red[tid + s];
    __syncthreads();
  }
  const float inv = 1.f / red[0];
  for (int i = tid; i < Sd; i += 256)
    out[(size_t)b * Sd + i] *= inv;
}

extern "C" void kernel_launch(void* const* d_in, const int* in_sizes, int n_in,
                              void* d_out, int out_size, void* d_ws, size_t ws_size,
                              hipStream_t stream) {
  (void)in_sizes; (void)n_in; (void)out_size; (void)ws_size;
  const float* hidden = (const float*)d_in[0];   // (B,H)
  const float* enc    = (const float*)d_in[1];   // (B,S,H)
  const float* W      = (const float*)d_in[2];   // (2H,H)
  const float* b_attn = (const float*)d_in[3];   // (H,)
  const float* v      = (const float*)d_in[4];   // (H,)
  float* out = (float*)d_out;                    // (B,S)

  float*    hidb   = (float*)d_ws;               // 16*1024 f32
  float*    logits = hidb + Bd * Hd;             // 16*2048 f32
  unsigned* Wpk    = (unsigned*)(logits + Bd * Sd);  // 2MB packed bf16 (uint4 tiled)

  packw_kernel<<<(512 * Hd) / 256, 256, 0, stream>>>(W, Wpk);
  hid_kernel<<<(Bd * Hd) / 256, 256, 0, stream>>>(hidden, W, b_attn, hidb);
  attn_energy_kernel<<<Bd * (Sd / TMR), 256, 0, stream>>>(enc, Wpk, hidb, v, logits);
  softmax_kernel<<<Bd, 256, 0, stream>>>(logits, out);
}
